// MultiHeadAttention_30279519437702
// MI455X (gfx1250) — matrix operand order repaired and checked by reading, not compiled, not run on hardware
//
#include <hip/hip_runtime.h>

typedef __attribute__((ext_vector_type(16))) _Float16 v16h;
typedef __attribute__((ext_vector_type(8)))  _Float16 v8h;
typedef __attribute__((ext_vector_type(4)))  _Float16 v4h;
typedef __attribute__((ext_vector_type(8)))  float    v8f;
typedef __attribute__((__vector_size__(4 * sizeof(int)))) int v4i;

#define BS 8
#define SEQ 1024
#define DMODEL 1024
#define NHEADS 16
#define DK 64

#define AS1 __attribute__((address_space(1)))
#define AS3 __attribute__((address_space(3)))

#if __has_builtin(__builtin_amdgcn_global_load_async_to_lds_b128) && \
    __has_builtin(__builtin_amdgcn_s_wait_asynccnt)
#define HAVE_ASYNC_LDS 1
__device__ __forceinline__ void async_cp16(const void* g, void* l) {
    // 16-byte global -> LDS copy, tracked by ASYNCcnt (no VGPR round trip).
    __builtin_amdgcn_global_load_async_to_lds_b128((AS1 v4i*)g, (AS3 v4i*)l, 0, 0);
}
__device__ __forceinline__ void async_wait0() { __builtin_amdgcn_s_wait_asynccnt(0); }
#else
#define HAVE_ASYNC_LDS 0
#endif

__device__ __forceinline__ v16h mk16(v8h a, v8h b) {
    v16h r;
#pragma unroll
    for (int i = 0; i < 8; ++i) { r[i] = a[i]; r[i + 8] = b[i]; }
    return r;
}

__device__ __forceinline__ v8f wmma_f16(v16h a, v16h b, v8f c) {
    // D = A(16x32 f16) * B(32x16 f16) + C(16x16 f32)
    return __builtin_amdgcn_wmma_f32_16x16x32_f16(false, a, false, b, (short)0, c,
                                                  false, false);
}

// ---------------------------------------------------------------------------
// Tiled GEMM:  Out(MxN) = A(MxK) * W(KxN) + bias   with M=8192, K=N=1024
// Block 128x256, 256 threads (8 waves); each wave owns a 64x64 patch:
// 16 accum tiles (128 VGPRs), 16 WMMAs per 32-wide K step, 16 ds_b128 loads.
// A staged row-major [m][k] f16; W staged transposed [n][k] f16.
// ---------------------------------------------------------------------------
template <bool A16MODE, bool SPLIT>
__global__ __launch_bounds__(256) void gemm_wmma(const void* __restrict__ Ap,
                                                 const float* __restrict__ W,
                                                 const float* __restrict__ bias,
                                                 void* __restrict__ Outp) {
    constexpr int BM = 128, BN = 256, BK = 32, LDA = 40;
    __shared__ _Float16 lA[BM * LDA];
    __shared__ _Float16 lB[BN * LDA];

    const int tid  = threadIdx.x;
    const int lane = tid & 31;
    const int wid  = tid >> 5;
    const int half = lane >> 4;
    const int l16  = lane & 15;
    const int bm   = blockIdx.y * BM;
    const int bn   = blockIdx.x * BN;
    const int wm   = (wid >> 2) * 64;   // 0 or 64
    const int wn   = (wid & 3) * 64;    // 0,64,128,192

    v8f acc[4][4];
#pragma unroll
    for (int i = 0; i < 4; ++i)
#pragma unroll
        for (int j = 0; j < 4; ++j) acc[i][j] = (v8f)0.0f;

    const int arow = tid >> 1;          // 0..127
    const int acol = (tid & 1) * 16;    // 0 or 16
    const int wk   = tid >> 3;          // 0..31
    const int wn0  = (tid & 7) * 32;    // 0..224

    for (int k0 = 0; k0 < DMODEL; k0 += BK) {
        // ---- stage A tile (convert f32->f16 if needed) ----
        if (A16MODE) {
            const _Float16* A = (const _Float16*)Ap;
            const _Float16* src = A + (size_t)(bm + arow) * DMODEL + k0 + acol;
#if HAVE_ASYNC_LDS
            async_cp16(src,     lA + arow * LDA + acol);
            async_cp16(src + 8, lA + arow * LDA + acol + 8);
#else
            *(v8h*)(lA + arow * LDA + acol)     = ((const v8h*)src)[0];
            *(v8h*)(lA + arow * LDA + acol + 8) = ((const v8h*)src)[1];
#endif
            if (k0 + BK < DMODEL) __builtin_prefetch(src + BK, 0, 1);
        } else {
            const float* A = (const float*)Ap;
            const float* src = A + (size_t)(bm + arow) * DMODEL + k0 + acol;
#pragma unroll
            for (int i = 0; i < 4; ++i) {
                float4 f = ((const float4*)src)[i];
                v4h hv;
                hv[0] = (_Float16)f.x; hv[1] = (_Float16)f.y;
                hv[2] = (_Float16)f.z; hv[3] = (_Float16)f.w;
                *(v4h*)(lA + arow * LDA + acol + 4 * i) = hv;
            }
            if (k0 + BK < DMODEL) __builtin_prefetch(src + BK, 0, 1);
        }
        // ---- stage W tile transposed: lB[n][k] ----
        {
            const float* wsrc = W + (size_t)(k0 + wk) * DMODEL + bn + wn0;
#pragma unroll
            for (int i = 0; i < 8; ++i) {
                float4 f = ((const float4*)wsrc)[i];
                lB[(wn0 + 4 * i + 0) * LDA + wk] = (_Float16)f.x;
                lB[(wn0 + 4 * i + 1) * LDA + wk] = (_Float16)f.y;
                lB[(wn0 + 4 * i + 2) * LDA + wk] = (_Float16)f.z;
                lB[(wn0 + 4 * i + 3) * LDA + wk] = (_Float16)f.w;
            }
            if (k0 + BK < DMODEL) __builtin_prefetch(wsrc + BK * DMODEL, 0, 1);
        }
#if HAVE_ASYNC_LDS
        if (A16MODE) async_wait0();
#endif
        __syncthreads();

        // ---- fragments + 16 WMMAs ----
        v16h afrag[4];
#pragma unroll
        for (int i = 0; i < 4; ++i) {
            const _Float16* base = lA + (wm + i * 16 + l16) * LDA;
            v8h r0 = *(const v8h*)(base + 8 * half);
            v8h r1 = *(const v8h*)(base + 16 + 8 * half);
            afrag[i] = mk16(r0, r1);
        }
        v16h bfrag[4];
#pragma unroll
        for (int j = 0; j < 4; ++j) {
            const _Float16* base = lB + (wn + j * 16 + l16) * LDA + 8 * half;
            bfrag[j] = mk16(*(const v8h*)base, *(const v8h*)(base + 16));
        }
#pragma unroll
        for (int i = 0; i < 4; ++i)
#pragma unroll
            for (int j = 0; j < 4; ++j) acc[i][j] = wmma_f16(afrag[i], bfrag[j], acc[i][j]);
        __syncthreads();
    }

    // ---- epilogue: bias + store ----
#pragma unroll
    for (int i = 0; i < 4; ++i)
#pragma unroll
        for (int j = 0; j < 4; ++j) {
            const int colg = bn + wn + j * 16 + l16;
            const float bv = bias[colg];
#pragma unroll
            for (int r = 0; r < 8; ++r) {
                const int rowg = bm + wm + i * 16 + r + 8 * half;
                const float val = acc[i][j][r] + bv;
                if (SPLIT) {
                    const int b = rowg >> 10, s = rowg & 1023;
                    const int h = colg >> 6,  d = colg & 63;
                    ((_Float16*)Outp)[((((size_t)b * NHEADS + h) * SEQ + s) * DK) + d] =
                        (_Float16)val;
                } else {
                    ((float*)Outp)[(size_t)rowg * DMODEL + colg] = val;
                }
            }
        }
}

// ---------------------------------------------------------------------------
// Flash attention per (b,h): computes S^T = K * Q^T so softmax statistics are
// per-lane (query on lane, keys in registers): 1 shfl_xor(16) per stat.
// Mask folded in via a single __ballot per key block. Block = 8 waves; each
// wave owns 16 query rows. K tile staged via async global->LDS when available.
// ---------------------------------------------------------------------------
__global__ __launch_bounds__(256) void attn_flash(const _Float16* __restrict__ Qh,
                                                  const _Float16* __restrict__ Kh,
                                                  const _Float16* __restrict__ Vh,
                                                  const int* __restrict__ mask,
                                                  _Float16* __restrict__ attn_out) {
    constexpr int LKV = 72;  // K tile stride:  [32 keys][64 d + pad]
    constexpr int LVT = 40;  // V^T stride:     [64 d][32 keys + pad]
    constexpr int LP  = 40;  // P scratch:      per wave [16 q][32 keys + pad]
    __shared__ _Float16 lK[32 * LKV];
    __shared__ _Float16 lVT[64 * LVT];
    __shared__ _Float16 lP[8 * 16 * LP];

    const int tid  = threadIdx.x;
    const int lane = tid & 31;
    const int wid  = tid >> 5;
    const int half = lane >> 4;
    const int l16  = lane & 15;
    const int bh   = blockIdx.y;
    const int b    = bh >> 4;
    const int h    = bh & 15;
    const int q0   = blockIdx.x * 128 + wid * 16;

    const size_t headoff = (size_t)bh * SEQ * DK;
    const _Float16* Q  = Qh + headoff;
    const _Float16* Kp = Kh + headoff;
    const _Float16* Vp = Vh + headoff;

    // Q^T fragments (B-layout: query on lane, d-contraction in table order: 8*half + e, 16 + 8*half + e).
    v16h qf[2];
#pragma unroll
    for (int c = 0; c < 2; ++c) {
        const _Float16* qb = Q + (size_t)(q0 + l16) * DK + 32 * c + 8 * half;
        qf[c] = mk16(*(const v8h*)qb, *(const v8h*)(qb + 16));
    }

    v8f o[4];
#pragma unroll
    for (int t = 0; t < 4; ++t) o[t] = (v8f)0.0f;
    float m = -3.0e38f;   // running max for query (lane & 15)
    float l = 0.0f;       // running sum for query (lane & 15)

    const int kr = tid >> 3;         // 0..31 : key row for coop load
    const int kc = (tid & 7) * 8;    // 0..56 : d offset

    for (int kb = 0; kb < SEQ; kb += 32) {
        // ---- cooperative K / V^T staging ----
        {
#if HAVE_ASYNC_LDS
            async_cp16(Kp + (size_t)(kb + kr) * DK + kc, lK + kr * LKV + kc);
#else
            v8h x = *(const v8h*)(Kp + (size_t)(kb + kr) * DK + kc);
            *(v8h*)(lK + kr * LKV + kc) = x;
#endif
            v8h y = *(const v8h*)(Vp + (size_t)(kb + kr) * DK + kc);
#pragma unroll
            for (int j = 0; j < 8; ++j) lVT[(kc + j) * LVT + kr] = y[j];
        }
#if HAVE_ASYNC_LDS
        async_wait0();
#endif
        __syncthreads();

        // ---- S^T = K * Q^T (two 16-key tiles as M, queries as N) ----
        v8f s[2];
        s[0] = (v8f)0.0f; s[1] = (v8f)0.0f;
#pragma unroll
        for (int n = 0; n < 2; ++n)
#pragma unroll
            for (int c = 0; c < 2; ++c) {
                const _Float16* kbse = lK + (16 * n + l16) * LKV + 32 * c;
                v8h r0 = *(const v8h*)(kbse + 8 * half);
                v8h r1 = *(const v8h*)(kbse + 16 + 8 * half);
                s[n] = wmma_f16(mk16(r0, r1), qf[c], s[n]);
            }

        // ---- mask bits: one ballot covers the 32 keys of this block ----
        const unsigned bits =
            (unsigned)__ballot(mask[b * SEQ + kb + lane] != 0);

        // ---- per-lane online softmax over keys in registers ----
        float sv[2][8];
        float tmax = -3.0e38f;
#pragma unroll
        for (int n = 0; n < 2; ++n)
#pragma unroll
            for (int r = 0; r < 8; ++r) {
                const int j = 16 * n + r + 8 * half;   // key index in block
                const float madd = ((bits >> j) & 1u) ? 0.0f : -1.0e9f;
                const float v = s[n][r] * 0.125f + madd;
                sv[n][r] = v;
                tmax = fmaxf(tmax, v);
            }
        tmax = fmaxf(tmax, __shfl_xor(tmax, 16, 32));
        const float mn    = fmaxf(m, tmax);
        const float alpha = __expf(m - mn);
        m = mn;

        _Float16* pp = lP + wid * 16 * LP;   // wave-private [16 q][32 keys]
        float rs = 0.0f;
#pragma unroll
        for (int n = 0; n < 2; ++n)
#pragma unroll
            for (int r = 0; r < 8; ++r) {
                const float p = __expf(sv[n][r] - mn);
                rs += p;
                pp[l16 * LP + 16 * n + r + 8 * half] = (_Float16)p;
            }
        rs += __shfl_xor(rs, 16, 32);
        l = l * alpha + rs;

        // ---- rescale O: broadcast alpha from query lanes to row registers --
        float af[8];
#pragma unroll
        for (int r = 0; r < 8; ++r) af[r] = __shfl(alpha, r + 8 * half, 16);
#pragma unroll
        for (int t = 0; t < 4; ++t)
#pragma unroll
            for (int r = 0; r < 8; ++r) o[t][r] *= af[r];

        // ---- reload P as A fragment (same-wave LDS, in-order) ----
        v16h pf;
        {
            const _Float16* base = pp + l16 * LP;
            v8h r0 = *(const v8h*)(base + 8 * half);
            v8h r1 = *(const v8h*)(base + 16 + 8 * half);
            pf = mk16(r0, r1);
        }

        // ---- O += P * V ----
#pragma unroll
        for (int t = 0; t < 4; ++t) {
            const _Float16* vb = lVT + (t * 16 + l16) * LVT + 8 * half;
            v16h vf = mk16(*(const v8h*)vb, *(const v8h*)(vb + 16));
            o[t] = wmma_f16(pf, vf, o[t]);
        }
        __syncthreads();
    }

    // ---- normalize (broadcast 1/l per row register) and store f16 concat ----
    const float inv = 1.0f / l;
    float ir[8];
#pragma unroll
    for (int r = 0; r < 8; ++r) ir[r] = __shfl(inv, r + 8 * half, 16);
#pragma unroll
    for (int r = 0; r < 8; ++r) {
        const int q = q0 + r + 8 * half;
#pragma unroll
        for (int t = 0; t < 4; ++t) {
            const int d = t * 16 + l16;
            attn_out[((size_t)b * SEQ + q) * DMODEL + h * DK + d] =
                (_Float16)(o[t][r] * ir[r]);
        }
    }
}

extern "C" void kernel_launch(void* const* d_in, const int* in_sizes, int n_in,
                              void* d_out, int out_size, void* d_ws, size_t ws_size,
                              hipStream_t stream) {
    const float* q    = (const float*)d_in[0];
    const float* k    = (const float*)d_in[1];
    const float* v    = (const float*)d_in[2];
    const int*   mask = (const int*)d_in[3];
    const float* Wq   = (const float*)d_in[4];
    const float* bq   = (const float*)d_in[5];
    const float* Wk   = (const float*)d_in[6];
    const float* bk   = (const float*)d_in[7];
    const float* Wv   = (const float*)d_in[8];
    const float* bv   = (const float*)d_in[9];
    const float* Wo   = (const float*)d_in[10];
    const float* bo   = (const float*)d_in[11];
    float* out = (float*)d_out;

    const size_t per = (size_t)BS * NHEADS * SEQ * DK;  // 8M f16 elements
    _Float16* qh   = (_Float16*)d_ws;
    _Float16* kh   = qh + per;
    _Float16* vh   = kh + per;
    _Float16* attn = vh + per;  // [B][S][DMODEL] f16

    dim3 gridG(DMODEL / 256, (BS * SEQ) / 128);  // (4, 64)
    dim3 blk(256);

    gemm_wmma<false, true><<<gridG, blk, 0, stream>>>((const void*)q, Wq, bq, (void*)qh);
    gemm_wmma<false, true><<<gridG, blk, 0, stream>>>((const void*)k, Wk, bk, (void*)kh);
    gemm_wmma<false, true><<<gridG, blk, 0, stream>>>((const void*)v, Wv, bv, (void*)vh);

    attn_flash<<<dim3(SEQ / 128, BS * NHEADS), blk, 0, stream>>>(qh, kh, vh, mask, attn);

    gemm_wmma<true, false><<<gridG, blk, 0, stream>>>((const void*)attn, Wo, bo, (void*)out);
}
